// TransformerBlock_74294344286708
// MI455X (gfx1250) — matrix operand order repaired and checked by reading, not compiled, not run on hardware
//
#include <hip/hip_runtime.h>
#include <hip/hip_bf16.h>

// ---------------------------------------------------------------------------
// Problem constants (B=2, C=192, H=W=256, heads=6, window=8, hd=32)
// ---------------------------------------------------------------------------
#define C_CH   192
#define HDIM   256
#define WDIM   256
#define BATCH  2
#define NHEAD  6
#define HD     32
#define NPIX   131072      // B*H*W
#define NELEM  25165824    // B*C*H*W
#define NWIN   2048        // B*(H/8)*(W/8)

typedef __attribute__((ext_vector_type(16))) __bf16 v16bf;
typedef __attribute__((ext_vector_type(8)))  float  v8f;

union FragU { v16bf v; uint4 u[2]; };

// ---------------------------------------------------------------------------
// Workspace layout (units of float; regions aliased across phases)
//   Phase A: S1 / L2 / H2 complex spectra
//   Phase B: hp (over dead S1), small block (over dead S1 tail)
//   Phase C: xmod/xn/attn (over dead L2), Q/K/VT (over dead H2), VF (over hp)
// ---------------------------------------------------------------------------
static const size_t OFF_S1    = 0;            // 50331648 f32 (complex)
static const size_t OFF_L2    = 50331648;     // 50331648 f32
static const size_t OFF_H2    = 100663296;    // 50331648 f32
static const size_t OFF_HP    = 0;            // 25165824 f32 NHWC   (S1 dead)
static const size_t OFF_LPSUM = 25165824;     // 384
static const size_t OFF_LPMAX = 25166208;     // 384
static const size_t OFF_W1    = 25166592;     // 384
static const size_t OFF_SM    = 25166976;     // 262144  [B,2,H,W]
static const size_t OFF_W2    = 25429120;     // 131072  [B,H,W]
static const size_t OFF_BIAS  = 25560192;     // 24576   [6,64,64]
static const size_t OFF_WQ    = 25584768;     // bf16 36864 (18432 slots)
static const size_t OFF_WK    = 25603200;
static const size_t OFF_WV    = 25621632;
static const size_t OFF_WP    = 25640064;     // ends 25658496
static const size_t OFF_XMOD  = 50331648;     // 25165824 f32 NHWC  (L2 dead)
static const size_t OFF_XN    = 75497472;     // bf16 [M,192] (12582912 slots)
static const size_t OFF_ATTN  = 88080384;     // bf16 [M,192]
static const size_t OFF_Q     = 100663296;    // bf16 [M,192]       (H2 dead)
static const size_t OFF_K     = 113246208;    // bf16 [M,192]; reused as YSUM
static const size_t OFF_VT    = 125829120;    // bf16 [win][h][d][t]
static const size_t OFF_VF    = 0;            // f32 NHWC  (hp dead after Q gemm)
static const size_t WS_NEED_BYTES = 138412032ull * 4ull;

// ---------------------------------------------------------------------------
// helpers
// ---------------------------------------------------------------------------
__device__ inline float warpSumF(float v){
  #pragma unroll
  for (int d = 16; d > 0; d >>= 1) v += __shfl_xor(v, d, 32);
  return v;
}
__device__ inline float warpMaxF(float v){
  #pragma unroll
  for (int d = 16; d > 0; d >>= 1) v = fmaxf(v, __shfl_xor(v, d, 32));
  return v;
}

// Stockham radix-2 FFT, N=256, 64 threads, ping-pong a<->b; result lands in a.
__device__ void fft256(float2* a, float2* b, float sgn){
  const int tid = threadIdx.x;
  int Ns = 1;
  float2 *src = a, *dst = b;
  #pragma unroll 1
  for (int stage = 0; stage < 8; ++stage) {
    __syncthreads();
    #pragma unroll
    for (int jj = 0; jj < 2; ++jj) {
      int j = tid + jj * 64;               // 0..127
      float2 v0 = src[j];
      float2 v1 = src[j + 128];
      float ang = sgn * 6.2831853071795864769f * (float)(j % Ns) / (float)(Ns * 2);
      float s, c;
      __sincosf(ang, &s, &c);
      float2 w1 = make_float2(v1.x * c - v1.y * s, v1.x * s + v1.y * c);
      int idxD = (j / Ns) * Ns * 2 + (j % Ns);
      dst[idxD]      = make_float2(v0.x + w1.x, v0.y + w1.y);
      dst[idxD + Ns] = make_float2(v0.x - w1.x, v0.y - w1.y);
    }
    Ns <<= 1;
    float2* t = src; src = dst; dst = t;
  }
  __syncthreads();
}

// ---------------------------------------------------------------------------
// K1: forward FFT along W.  grid = B*C*H (98304), block 64
// ---------------------------------------------------------------------------
__global__ void k_rowfft(const float* __restrict__ x, float* __restrict__ S1){
  __shared__ float2 a[256], b[256];
  const size_t base = (size_t)blockIdx.x * 256;
  for (int i = threadIdx.x; i < 256; i += 64) a[i] = make_float2(x[base + i], 0.f);
  fft256(a, b, -1.f);
  for (int i = threadIdx.x; i < 256; i += 64) {
    S1[2*(base+i)]   = a[i].x;
    S1[2*(base+i)+1] = a[i].y;
  }
}

// ---------------------------------------------------------------------------
// K2: column FFT, Gaussian split, column IFFT (both branches).
// grid = B*C*W (98304), block 64
// ---------------------------------------------------------------------------
__global__ void k_colfft_filter(const float* __restrict__ S1,
                                float* __restrict__ L2, float* __restrict__ H2){
  __shared__ float2 a[256], b[256], spec[256];
  const int bc = blockIdx.x >> 8;
  const int kx = blockIdx.x & 255;
  const size_t pb = (size_t)bc * 65536;
  for (int i = threadIdx.x; i < 256; i += 64) {
    size_t idx = 2*(pb + (size_t)i*256 + kx);
    a[i] = make_float2(S1[idx], S1[idx+1]);
  }
  fft256(a, b, -1.f);
  for (int i = threadIdx.x; i < 256; i += 64) spec[i] = a[i];
  __syncthreads();
  const float step = 256.f / 255.f;
  const float Xc = -128.f + (float)((kx + 128) & 255) * step;
  const float inv2c2 = 1.f / (2.f * 25.6f * 25.6f);
  // low-pass branch
  for (int i = threadIdx.x; i < 256; i += 64) {
    float Yc = -128.f + (float)((i + 128) & 255) * step;
    float g = __expf(-(Xc*Xc + Yc*Yc) * inv2c2);
    a[i] = make_float2(spec[i].x * g, spec[i].y * g);
  }
  fft256(a, b, 1.f);
  for (int i = threadIdx.x; i < 256; i += 64) {
    size_t idx = 2*(pb + (size_t)i*256 + kx);
    L2[idx]   = a[i].x * (1.f/256.f);
    L2[idx+1] = a[i].y * (1.f/256.f);
  }
  __syncthreads();
  // high-pass branch
  for (int i = threadIdx.x; i < 256; i += 64) {
    float Yc = -128.f + (float)((i + 128) & 255) * step;
    float g = __expf(-(Xc*Xc + Yc*Yc) * inv2c2);
    a[i] = make_float2(spec[i].x * (1.f - g), spec[i].y * (1.f - g));
  }
  fft256(a, b, 1.f);
  for (int i = threadIdx.x; i < 256; i += 64) {
    size_t idx = 2*(pb + (size_t)i*256 + kx);
    H2[idx]   = a[i].x * (1.f/256.f);
    H2[idx+1] = a[i].y * (1.f/256.f);
  }
}

// ---------------------------------------------------------------------------
// K3a: row IFFT + |.|  -> hp in NHWC.  grid = B*C*H, block 64
// ---------------------------------------------------------------------------
__global__ void k_rowifft_hp(const float* __restrict__ H2, float* __restrict__ hp){
  __shared__ float2 a[256], b[256];
  const int row = blockIdx.x;
  const size_t base = (size_t)row * 256;
  for (int i = threadIdx.x; i < 256; i += 64)
    a[i] = make_float2(H2[2*(base+i)], H2[2*(base+i)+1]);
  fft256(a, b, 1.f);
  const int h = row & 255;
  const int c = (row >> 8) % C_CH;
  const int bb = row / (C_CH * 256);
  for (int i = threadIdx.x; i < 256; i += 64) {
    float re = a[i].x * (1.f/256.f), im = a[i].y * (1.f/256.f);
    hp[((((size_t)bb*256 + h)*256) + i)*C_CH + c] = sqrtf(re*re + im*im);
  }
}

// K3b: row IFFT + |.| for low-pass, fused per-(b,c) mean/max reduction.
__global__ void k_rowifft_lp(const float* __restrict__ L2,
                             float* __restrict__ lpsum, float* __restrict__ lpmax){
  __shared__ float2 a[256], b[256];
  __shared__ float rs[2], rm[2];
  const int row = blockIdx.x;
  const size_t base = (size_t)row * 256;
  for (int i = threadIdx.x; i < 256; i += 64)
    a[i] = make_float2(L2[2*(base+i)], L2[2*(base+i)+1]);
  fft256(a, b, 1.f);
  float s = 0.f, m = 0.f;
  for (int i = threadIdx.x; i < 256; i += 64) {
    float re = a[i].x * (1.f/256.f), im = a[i].y * (1.f/256.f);
    float mag = sqrtf(re*re + im*im);
    s += mag; m = fmaxf(m, mag);
  }
  s = warpSumF(s); m = warpMaxF(m);
  const int lane = threadIdx.x & 31, wid = threadIdx.x >> 5;
  if (lane == 0) { rs[wid] = s; rm[wid] = m; }
  __syncthreads();
  if (threadIdx.x == 0) {
    const int c = (row >> 8) % C_CH;
    const int bb = row / (C_CH * 256);
    atomicAdd(&lpsum[bb*C_CH + c], rs[0] + rs[1]);
    atomicMax((unsigned int*)&lpmax[bb*C_CH + c], __float_as_uint(fmaxf(rm[0], rm[1])));
  }
}

// ---------------------------------------------------------------------------
// K4: channel attention  (2 blocks x 192 threads)
// ---------------------------------------------------------------------------
__global__ void k_chanatt(const float* __restrict__ lpsum, const float* __restrict__ lpmax,
                          const float* __restrict__ caw1, const float* __restrict__ caw2,
                          float* __restrict__ w1o){
  const int bb = blockIdx.x, c = threadIdx.x;
  __shared__ float hm[12], hM[12];
  if (c < 12) {
    float am = 0.f, aM = 0.f;
    for (int k = 0; k < C_CH; ++k) {
      float wv = caw1[c*C_CH + k];
      am += wv * (lpsum[bb*C_CH + k] * (1.f/65536.f));
      aM += wv * lpmax[bb*C_CH + k];
    }
    hm[c] = fmaxf(am, 0.f);
    hM[c] = fmaxf(aM, 0.f);
  }
  __syncthreads();
  float o = 0.f;
  for (int j = 0; j < 12; ++j) o += caw2[c*12 + j] * (hm[j] + hM[j]);
  w1o[bb*C_CH + c] = 1.f / (1.f + __expf(-o));
}

// K5a: sm = [mean_c, max_c](hp).  one wave per pixel, block 256, grid 16384
__global__ void k_sm(const float* __restrict__ hp, float* __restrict__ sm){
  const int wid = threadIdx.x >> 5, lane = threadIdx.x & 31;
  const size_t p = (size_t)blockIdx.x * 8 + wid;
  const float* row = hp + p * C_CH;
  float s = 0.f, m = -1e30f;
  #pragma unroll
  for (int k = 0; k < 6; ++k) { float v = row[lane + 32*k]; s += v; m = fmaxf(m, v); }
  s = warpSumF(s); m = warpMaxF(m);
  if (lane == 0) {
    size_t bb = p >> 16, pix = p & 65535;
    sm[bb*131072 + pix]         = s * (1.f / (float)C_CH);
    sm[bb*131072 + 65536 + pix] = m;
  }
}

// K5b: combined 3/5/7 conv + sigmoid -> w2.  block 256, grid 512
__global__ void k_spatial(const float* __restrict__ sm,
                          const float* __restrict__ w3, const float* __restrict__ w5,
                          const float* __restrict__ w7, float* __restrict__ w2o){
  __shared__ float wt[2][49];
  const int t = threadIdx.x;
  if (t < 98) {
    int ch = t / 49, k = t % 49, dy = k/7 - 3, dx = k%7 - 3;
    float v = w7[ch*49 + k];
    if (abs(dy) <= 2 && abs(dx) <= 2) v += w5[ch*25 + (dy+2)*5 + (dx+2)];
    if (abs(dy) <= 1 && abs(dx) <= 1) v += w3[ch*9  + (dy+1)*3 + (dx+1)];
    wt[ch][k] = v;
  }
  __syncthreads();
  const size_t p = (size_t)blockIdx.x * 256 + threadIdx.x;   // exactly 131072
  const int bb = (int)(p >> 16), y = (int)((p >> 8) & 255), x = (int)(p & 255);
  float acc = 0.f;
  for (int dy = -3; dy <= 3; ++dy) {
    int yy = y + dy; if (yy < 0 || yy > 255) continue;
    for (int dx = -3; dx <= 3; ++dx) {
      int xx = x + dx; if (xx < 0 || xx > 255) continue;
      int k = (dy+3)*7 + (dx+3);
      size_t base = (size_t)bb*131072 + yy*256 + xx;
      acc += wt[0][k]*sm[base] + wt[1][k]*sm[base + 65536];
    }
  }
  w2o[p] = 1.f / (1.f + __expf(-acc));
}

// K6: modulate x by w1,w2, NCHW -> NHWC (xmod is also the residual identity)
__global__ void k_modulate(const float* __restrict__ x, const float* __restrict__ w1,
                           const float* __restrict__ w2, float* __restrict__ xmod){
  const size_t i = (size_t)blockIdx.x * 256 + threadIdx.x;   // NCHW index
  const int xp = (int)(i & 255), y = (int)((i >> 8) & 255);
  const int c = (int)((i >> 16) % C_CH), bb = (int)(i / ((size_t)C_CH * 65536));
  float v = x[i] * w1[bb*C_CH + c] * w2[(size_t)bb*65536 + y*256 + xp];
  xmod[((((size_t)bb*256 + y)*256) + xp)*C_CH + c] = v;
}

// K7: layernorm over channels -> xn (bf16, NHWC).  wave per pixel
__global__ void k_ln(const float* __restrict__ xmod, const float* __restrict__ lnw,
                     const float* __restrict__ lnb, __bf16* __restrict__ xn){
  const int wid = threadIdx.x >> 5, lane = threadIdx.x & 31;
  const size_t p = (size_t)blockIdx.x * 8 + wid;
  const float* row = xmod + p * C_CH;
  float vals[6], s = 0.f, s2 = 0.f;
  #pragma unroll
  for (int k = 0; k < 6; ++k) { vals[k] = row[lane + 32*k]; s += vals[k]; s2 += vals[k]*vals[k]; }
  s = warpSumF(s); s2 = warpSumF(s2);
  float mu = s * (1.f / (float)C_CH);
  float inv = rsqrtf(s2 * (1.f / (float)C_CH) - mu*mu + 1e-5f);
  #pragma unroll
  for (int k = 0; k < 6; ++k) {
    int c = lane + 32*k;
    xn[p*C_CH + c] = (__bf16)((vals[k] - mu) * inv * lnw[c] + lnb[c]);
  }
}

// misc: f32 -> bf16 weight conversion; zero init
__global__ void k_cvtw(const float* __restrict__ a, __bf16* __restrict__ o, int n){
  int i = blockIdx.x * 256 + threadIdx.x;
  if (i < n) o[i] = (__bf16)a[i];
}
__global__ void k_zero(float* p, int n){
  int i = blockIdx.x * 256 + threadIdx.x;
  if (i < n) p[i] = 0.f;
}

// K8: relative-position bias table  biasT[h][i][j].  grid 64 x block 64
__global__ void k_bias(const float* __restrict__ w1m, const float* __restrict__ b1,
                       const float* __restrict__ w2m, const float* __restrict__ b2,
                       float* __restrict__ biasT){
  const int t = blockIdx.x * 64 + threadIdx.x;   // 4096 = i*64 + j
  const int n = t >> 6, m = t & 63;
  float ry = (float)(n/8 - m/8), rx = (float)(n%8 - m%8);
  ry = copysignf(log1pf(fabsf(ry)), ry);
  rx = copysignf(log1pf(fabsf(rx)), rx);
  float o[NHEAD];
  #pragma unroll
  for (int h = 0; h < NHEAD; ++h) o[h] = b2[h];
  for (int k = 0; k < 256; ++k) {
    float hk = fmaxf(w1m[k*2]*ry + w1m[k*2+1]*rx + b1[k], 0.f);
    #pragma unroll
    for (int h = 0; h < NHEAD; ++h) o[h] += w2m[h*256 + k] * hk;
  }
  #pragma unroll
  for (int h = 0; h < NHEAD; ++h) biasT[h*4096 + t] = o[h];
}

// ---------------------------------------------------------------------------
// K9: bf16 WMMA GEMM  Y[M,192] = A[M,192] @ W^T + epilogue
//   mode 0: bf16 out (+optional add_nhwc) * scale        (Q / K)
//   mode 1: f32 NHWC out                                 (V)
//   mode 2: f32 NCHW out + add_nhwc (residual)           (proj -> d_out)
// grid 1024, block 256 (8 waves; wave = one 16-row strip x 12 N-tiles)
// ---------------------------------------------------------------------------
__global__ void __launch_bounds__(256) k_gemm192(
    const __bf16* __restrict__ A, const __bf16* __restrict__ Wt,
    const float* __restrict__ bias, const float* __restrict__ add_nhwc,
    float scale, __bf16* __restrict__ outb, float* __restrict__ outf, int mode)
{
  const int wid = threadIdx.x >> 5, lane = threadIdx.x & 31;
  const int mt = blockIdx.x * 8 + wid;
  const int row0 = mt * 16;
  const int lrow = lane & 15;
  const bool hi = lane >= 16;

  v8f acc[12];
  #pragma unroll
  for (int nt = 0; nt < 12; ++nt)
    #pragma unroll
    for (int r = 0; r < 8; ++r) acc[nt][r] = 0.f;

  const __bf16* arow = A + (size_t)(row0 + lrow) * C_CH;
  #pragma unroll
  for (int ks = 0; ks < 6; ++ks) {
    FragU af;
    const __bf16* ap = arow + ks*32 + (hi ? 8 : 0);
    af.u[0] = *reinterpret_cast<const uint4*>(ap);
    af.u[1] = *reinterpret_cast<const uint4*>(ap + 16);
    #pragma unroll
    for (int nt = 0; nt < 12; ++nt) {
      FragU bw;
      const __bf16* bp = Wt + (size_t)(nt*16 + lrow) * C_CH + ks*32 + (hi ? 8 : 0);
      bw.u[0] = *reinterpret_cast<const uint4*>(bp);
      bw.u[1] = *reinterpret_cast<const uint4*>(bp + 16);
      acc[nt] = __builtin_amdgcn_wmma_f32_16x16x32_bf16(
          false, af.v, false, bw.v, (short)0, acc[nt], false, false);
    }
  }

  #pragma unroll
  for (int nt = 0; nt < 12; ++nt) {
    const int n = nt*16 + lrow;
    const float bv = bias[n];
    #pragma unroll
    for (int r = 0; r < 8; ++r) {
      const size_t p = (size_t)(row0 + r + (hi ? 8 : 0));
      float v = acc[nt][r] + bv;
      if (mode == 0) {
        if (add_nhwc) v += add_nhwc[p*C_CH + n];
        outb[p*C_CH + n] = (__bf16)(v * scale);
      } else if (mode == 1) {
        outf[p*C_CH + n] = v;
      } else {
        v += add_nhwc[p*C_CH + n];
        const size_t bb = p >> 16, y = (p >> 8) & 255, xx = p & 255;
        outf[((bb*C_CH + n)*256 + y)*256 + xx] = v;
      }
    }
  }
}

// K10: V (f32 NHWC) -> VT bf16 [win][head][dim][tok]
__global__ void k_vt(const float* __restrict__ VF, __bf16* __restrict__ VT){
  const size_t i = (size_t)blockIdx.x * 256 + threadIdx.x;   // NELEM total
  const int t = (int)(i & 63), d = (int)((i >> 6) & 31);
  const int h = (int)((i / 2048) % NHEAD);
  const int win = (int)(i / 12288);
  const int bb = win >> 10, wr = win & 1023;
  const int y = (wr >> 5) * 8 + (t >> 3), x = (wr & 31) * 8 + (t & 7);
  VT[i] = (__bf16)VF[((((size_t)bb*256 + y)*256) + x)*C_CH + h*HD + d];
}

// ---------------------------------------------------------------------------
// K11: window attention. block = window (192 thr = 6 waves), wave = head.
// ---------------------------------------------------------------------------
__global__ void __launch_bounds__(192) k_attn(
    const __bf16* __restrict__ Qb, const __bf16* __restrict__ Kb,
    const __bf16* __restrict__ VT, const float* __restrict__ biasT,
    __bf16* __restrict__ Ob)
{
  __shared__ __align__(16) __bf16 plds[NHEAD][64][72];
  const int h = threadIdx.x >> 5, lane = threadIdx.x & 31;
  const int lrow = lane & 15;
  const bool hi = lane >= 16;
  const int win = blockIdx.x;
  const int bb = win >> 10, wr = win & 1023;
  const int y0 = (wr >> 5) * 8, x0 = (wr & 31) * 8;

  auto tokbase = [&](int t) -> size_t {
    int y = y0 + (t >> 3), x = x0 + (t & 7);
    return ((((size_t)bb*256 + y)*256) + x)*C_CH + h*HD;
  };

  FragU QA[4], KBf[4];
  #pragma unroll
  for (int mt = 0; mt < 4; ++mt) {
    const __bf16* qp = Qb + tokbase(mt*16 + lrow) + (hi ? 8 : 0);
    QA[mt].u[0] = *reinterpret_cast<const uint4*>(qp);
    QA[mt].u[1] = *reinterpret_cast<const uint4*>(qp + 16);
  }
  #pragma unroll
  for (int nt = 0; nt < 4; ++nt) {
    const __bf16* kp = Kb + tokbase(nt*16 + lrow) + (hi ? 8 : 0);
    KBf[nt].u[0] = *reinterpret_cast<const uint4*>(kp);
    KBf[nt].u[1] = *reinterpret_cast<const uint4*>(kp + 16);
  }

  v8f S[4][4];
  #pragma unroll
  for (int mt = 0; mt < 4; ++mt)
    #pragma unroll
    for (int nt = 0; nt < 4; ++nt) {
      #pragma unroll
      for (int r = 0; r < 8; ++r) S[mt][nt][r] = 0.f;
      S[mt][nt] = __builtin_amdgcn_wmma_f32_16x16x32_bf16(
          false, QA[mt].v, false, KBf[nt].v, (short)0, S[mt][nt], false, false);
    }

  // + relative-position bias, then row softmax (rows live on 16-lane halves)
  const float* bt = biasT + h*4096;
  #pragma unroll
  for (int mt = 0; mt < 4; ++mt) {
    #pragma unroll
    for (int nt = 0; nt < 4; ++nt)
      #pragma unroll
      for (int r = 0; r < 8; ++r)
        S[mt][nt][r] += bt[(mt*16 + r + (hi ? 8 : 0))*64 + nt*16 + lrow];
    #pragma unroll
    for (int r = 0; r < 8; ++r) {
      float mx = -1e30f;
      #pragma unroll
      for (int nt = 0; nt < 4; ++nt) mx = fmaxf(mx, S[mt][nt][r]);
      #pragma unroll
      for (int d = 1; d <= 8; d <<= 1) mx = fmaxf(mx, __shfl_xor(mx, d, 32));
      float sum = 0.f;
      #pragma unroll
      for (int nt = 0; nt < 4; ++nt) {
        float e = __expf(S[mt][nt][r] - mx);
        S[mt][nt][r] = e; sum += e;
      }
      #pragma unroll
      for (int d = 1; d <= 8; d <<= 1) sum += __shfl_xor(sum, d, 32);
      float inv = 1.f / sum;
      #pragma unroll
      for (int nt = 0; nt < 4; ++nt) S[mt][nt][r] *= inv;
    }
  }

  // stage P in LDS (bf16) to re-stripe into A-fragments
  #pragma unroll
  for (int mt = 0; mt < 4; ++mt)
    #pragma unroll
    for (int nt = 0; nt < 4; ++nt)
      #pragma unroll
      for (int r = 0; r < 8; ++r)
        plds[h][mt*16 + r + (hi ? 8 : 0)][nt*16 + lrow] = (__bf16)S[mt][nt][r];
  __syncthreads();

  FragU VB[2][2];
  const size_t vbase = ((size_t)win*NHEAD + h) * 2048;   // 32*64
  #pragma unroll
  for (int kt = 0; kt < 2; ++kt)
    #pragma unroll
    for (int dt = 0; dt < 2; ++dt) {
      const __bf16* vp = VT + vbase + (size_t)(dt*16 + lrow)*64 + kt*32 + (hi ? 8 : 0);
      VB[kt][dt].u[0] = *reinterpret_cast<const uint4*>(vp);
      VB[kt][dt].u[1] = *reinterpret_cast<const uint4*>(vp + 16);
    }

  v8f O[4][2];
  #pragma unroll
  for (int mt = 0; mt < 4; ++mt)
    #pragma unroll
    for (int dt = 0; dt < 2; ++dt)
      #pragma unroll
      for (int r = 0; r < 8; ++r) O[mt][dt][r] = 0.f;

  #pragma unroll
  for (int mt = 0; mt < 4; ++mt)
    #pragma unroll
    for (int kt = 0; kt < 2; ++kt) {
      FragU PA;
      const __bf16* lp = &plds[h][mt*16 + lrow][kt*32 + (hi ? 8 : 0)];
      PA.u[0] = *reinterpret_cast<const uint4*>(lp);
      PA.u[1] = *reinterpret_cast<const uint4*>(lp + 16);
      #pragma unroll
      for (int dt = 0; dt < 2; ++dt)
        O[mt][dt] = __builtin_amdgcn_wmma_f32_16x16x32_bf16(
            false, PA.v, false, VB[kt][dt].v, (short)0, O[mt][dt], false, false);
    }

  #pragma unroll
  for (int mt = 0; mt < 4; ++mt)
    #pragma unroll
    for (int dt = 0; dt < 2; ++dt)
      #pragma unroll
      for (int r = 0; r < 8; ++r) {
        int t = mt*16 + r + (hi ? 8 : 0);
        Ob[tokbase(t) + dt*16 + lrow] = (__bf16)O[mt][dt][r];
      }
}

// K12: depthwise 5x5 (reflect pad) on V + attn -> ysum (bf16)
__global__ void k_dwconv(const float* __restrict__ VF, const __bf16* __restrict__ attn,
                         const float* __restrict__ dww, const float* __restrict__ dwb,
                         __bf16* __restrict__ ysum){
  const size_t i = (size_t)blockIdx.x * 256 + threadIdx.x;   // p*192 + c
  const int c = (int)(i % C_CH);
  const size_t p = i / C_CH;
  const int xx = (int)(p & 255), y = (int)((p >> 8) & 255), bb = (int)(p >> 16);
  float acc = dwb[c];
  #pragma unroll
  for (int dy = -2; dy <= 2; ++dy) {
    int ry = y + dy; ry = ry < 0 ? -ry : (ry > 255 ? 510 - ry : ry);
    #pragma unroll
    for (int dx = -2; dx <= 2; ++dx) {
      int rx = xx + dx; rx = rx < 0 ? -rx : (rx > 255 ? 510 - rx : rx);
      acc += dww[c*25 + (dy+2)*5 + (dx+2)] *
             VF[((((size_t)bb*256 + ry)*256) + rx)*C_CH + c];
    }
  }
  acc += (float)attn[i];
  ysum[i] = (__bf16)acc;
}

// ---------------------------------------------------------------------------
extern "C" void kernel_launch(void* const* d_in, const int* in_sizes, int n_in,
                              void* d_out, int out_size, void* d_ws, size_t ws_size,
                              hipStream_t stream){
  (void)in_sizes; (void)n_in; (void)out_size;
  if (ws_size < WS_NEED_BYTES) return;

  const float* x      = (const float*)d_in[0];
  const float* ln_w   = (const float*)d_in[1];
  const float* ln_b   = (const float*)d_in[2];
  const float* ca_w1  = (const float*)d_in[3];
  const float* ca_w2  = (const float*)d_in[4];
  const float* sa_w3  = (const float*)d_in[5];
  const float* sa_w5  = (const float*)d_in[6];
  const float* sa_w7  = (const float*)d_in[7];
  const float* q_w    = (const float*)d_in[8];
  const float* q_b    = (const float*)d_in[9];
  const float* k_w    = (const float*)d_in[10];
  const float* k_b    = (const float*)d_in[11];
  const float* v_w    = (const float*)d_in[12];
  const float* v_b    = (const float*)d_in[13];
  const float* proj_w = (const float*)d_in[14];
  const float* proj_b = (const float*)d_in[15];
  const float* mw1    = (const float*)d_in[16];
  const float* mb1    = (const float*)d_in[17];
  const float* mw2    = (const float*)d_in[18];
  const float* mb2    = (const float*)d_in[19];
  const float* dw_w   = (const float*)d_in[20];
  const float* dw_b   = (const float*)d_in[21];

  float* wsf = (float*)d_ws;
  float* S1   = wsf + OFF_S1;
  float* L2   = wsf + OFF_L2;
  float* H2   = wsf + OFF_H2;
  float* hp   = wsf + OFF_HP;
  float* lps  = wsf + OFF_LPSUM;
  float* lpm  = wsf + OFF_LPMAX;
  float* w1b  = wsf + OFF_W1;
  float* smb  = wsf + OFF_SM;
  float* w2b  = wsf + OFF_W2;
  float* biasT= wsf + OFF_BIAS;
  float* xmod = wsf + OFF_XMOD;
  float* VF   = wsf + OFF_VF;
  __bf16* wq  = (__bf16*)(wsf + OFF_WQ);
  __bf16* wk  = (__bf16*)(wsf + OFF_WK);
  __bf16* wv  = (__bf16*)(wsf + OFF_WV);
  __bf16* wp  = (__bf16*)(wsf + OFF_WP);
  __bf16* xn  = (__bf16*)(wsf + OFF_XN);
  __bf16* atb = (__bf16*)(wsf + OFF_ATTN);
  __bf16* Qb  = (__bf16*)(wsf + OFF_Q);
  __bf16* Kb  = (__bf16*)(wsf + OFF_K);
  __bf16* VT  = (__bf16*)(wsf + OFF_VT);
  __bf16* ysum= (__bf16*)(wsf + OFF_K);   // reuses Kb after attention

  // --- frequency split ---
  k_rowfft<<<98304, 64, 0, stream>>>(x, S1);
  k_colfft_filter<<<98304, 64, 0, stream>>>(S1, L2, H2);
  // S1 dead: its tail hosts small buffers / bf16 weights / bias table
  k_zero<<<3, 256, 0, stream>>>(lps, 768);
  k_cvtw<<<144, 256, 0, stream>>>(q_w, wq, 36864);
  k_cvtw<<<144, 256, 0, stream>>>(k_w, wk, 36864);
  k_cvtw<<<144, 256, 0, stream>>>(v_w, wv, 36864);
  k_cvtw<<<144, 256, 0, stream>>>(proj_w, wp, 36864);
  k_bias<<<64, 64, 0, stream>>>(mw1, mb1, mw2, mb2, biasT);
  k_rowifft_hp<<<98304, 64, 0, stream>>>(H2, hp);
  k_rowifft_lp<<<98304, 64, 0, stream>>>(L2, lps, lpm);
  // --- channel / spatial attention, modulation ---
  k_chanatt<<<2, 192, 0, stream>>>(lps, lpm, ca_w1, ca_w2, w1b);
  k_sm<<<16384, 256, 0, stream>>>(hp, smb);
  k_spatial<<<512, 256, 0, stream>>>(smb, sa_w3, sa_w5, sa_w7, w2b);
  k_modulate<<<98304, 256, 0, stream>>>(x, w1b, w2b, xmod);
  k_ln<<<16384, 256, 0, stream>>>(xmod, ln_w, ln_b, xn);
  // --- QKV projections (bf16 WMMA) ---
  k_gemm192<<<1024, 256, 0, stream>>>(xn, wq, q_b, hp, 0.17677669529663687f,
                                      Qb, nullptr, 0);         // Q (+prompt, *hd^-0.5)
  k_gemm192<<<1024, 256, 0, stream>>>(xn, wk, k_b, nullptr, 1.f,
                                      Kb, nullptr, 0);         // K
  k_gemm192<<<1024, 256, 0, stream>>>(xn, wv, v_b, nullptr, 1.f,
                                      nullptr, VF, 1);         // V f32 (over dead hp)
  k_vt<<<98304, 256, 0, stream>>>(VF, VT);
  // --- window attention ---
  k_attn<<<NWIN, 192, 0, stream>>>(Qb, Kb, VT, biasT, atb);
  // --- depthwise conv + merge ---
  k_dwconv<<<98304, 256, 0, stream>>>(VF, atb, dw_w, dw_b, ysum);
  // --- output projection + residual (writes NCHW d_out) ---
  k_gemm192<<<1024, 256, 0, stream>>>(ysum, wp, proj_b, xmod, 1.f,
                                      nullptr, (float*)d_out, 2);
}
